// SelfAttention_71597104824389
// MI455X (gfx1250) — matrix operand order repaired and checked by reading, not compiled, not run on hardware
//
#include <hip/hip_runtime.h>

// ---------------- problem constants ----------------
#define HEADS  8
#define DHEAD  64
#define SEQ    2048
#define BATCH  4
#define DMODEL 512            // HEADS * DHEAD
#define D3     1536           // 3 * DMODEL
static constexpr float ATT_SCALE = 0.125f;   // DHEAD^-0.5

// ---------------- WMMA types ----------------
typedef __attribute__((ext_vector_type(16))) __bf16 v16bf;
typedef __attribute__((ext_vector_type(8)))  float  v8f;
typedef __attribute__((ext_vector_type(4)))  unsigned u32x4;
typedef __attribute__((ext_vector_type(8)))  unsigned u32x8;

struct U8x32 { uint4 lo, hi; };                 // 32 bytes -> one v16bf
struct H16   { unsigned short h[16]; };         // 32 bytes -> one v16bf

// fp32 -> bf16 round-to-nearest-even
__device__ __forceinline__ unsigned short f2bf(float f) {
  unsigned u = __builtin_bit_cast(unsigned, f);
  u += 0x7FFFu + ((u >> 16) & 1u);
  return (unsigned short)(u >> 16);
}
__device__ __forceinline__ unsigned pack2bf(float lo, float hi) {
  return (unsigned)f2bf(lo) | ((unsigned)f2bf(hi) << 16);
}

// load a v16bf whose 16 elements are two contiguous 16B runs (A-fragment style)
__device__ __forceinline__ v16bf load_frag2(const unsigned short* p0,
                                            const unsigned short* p1) {
  U8x32 u;
  u.lo = *(const uint4*)p0;
  u.hi = *(const uint4*)p1;
  return __builtin_bit_cast(v16bf, u);
}
// load a v16bf from 32 contiguous bytes (B-fragment style)
__device__ __forceinline__ v16bf load_frag1(const unsigned short* p) {
  return load_frag2(p, p + 8);
}
// convert 8 contiguous fp32 to 8 bf16
__device__ __forceinline__ void cvt8(const float* p, unsigned short* h) {
  float4 a = *(const float4*)p;
  float4 b = *(const float4*)(p + 4);
  h[0]=f2bf(a.x); h[1]=f2bf(a.y); h[2]=f2bf(a.z); h[3]=f2bf(a.w);
  h[4]=f2bf(b.x); h[5]=f2bf(b.y); h[6]=f2bf(b.z); h[7]=f2bf(b.w);
}
// truncate generic pointer -> LDS byte offset (aperture keeps offset in [31:0])
__device__ __forceinline__ unsigned lds_off(const void* p) {
  return (unsigned)(unsigned long long)p;
}

#define WMMA_BF16(A,B,C) \
  __builtin_amdgcn_wmma_f32_16x16x32_bf16(false, (A), false, (B), (short)0, (C), false, false)

// =====================================================================
// Kernel 1: qkv = x @ w_qkv  (fp32 in, bf16 out, split into q/k/v [B][H][N][64])
// block = 256 thr = 8 waves (4 x 2); block tile 64(M) x 128(N); K step 32
// =====================================================================
__global__ __launch_bounds__(256) void qkv_gemm(const float* __restrict__ x,
                                                const float* __restrict__ wqkv,
                                                unsigned short* __restrict__ qws,
                                                unsigned short* __restrict__ kws,
                                                unsigned short* __restrict__ vws) {
  __shared__ __align__(16) unsigned short bt[128 * 40];   // w_qkv tile, [n][k] bf16
  const int tid  = threadIdx.x;
  const int w    = tid >> 5, l = tid & 31, half = l >> 4, ln = l & 15;
  const int wm   = w >> 1, wn = w & 1;
  const int m0   = blockIdx.x * 64;
  const int n0   = blockIdx.y * 128;
  const float* aptr = x + (size_t)(m0 + wm * 16 + ln) * DMODEL;

  v8f c[4] = {};
  for (int kc = 0; kc < DMODEL; kc += 32) {
    __syncthreads();
#pragma unroll
    for (int i = 0; i < 8; i++) {                     // stage 32x128 weights, transposed
      int idx = tid + i * 256;                        // k-pair major
      int kp = idx >> 7, n = idx & 127;
      float a0 = wqkv[(size_t)(kc + 2 * kp)     * D3 + n0 + n];
      float a1 = wqkv[(size_t)(kc + 2 * kp + 1) * D3 + n0 + n];
      *(unsigned*)&bt[n * 40 + 2 * kp] = pack2bf(a0, a1);
    }
    if (kc + 32 < DMODEL)
      __builtin_prefetch(wqkv + (size_t)(kc + 32) * D3 + n0 + (tid & 127), 0, 0);
    __syncthreads();

    H16 hh;
    cvt8(aptr + kc +      8 * half, hh.h);            // A elements 0..7
    cvt8(aptr + kc + 16 + 8 * half, hh.h + 8);        // A elements 8..15
    v16bf a = __builtin_bit_cast(v16bf, hh);
#pragma unroll
    for (int nt = 0; nt < 4; nt++) {
      v16bf b = load_frag2(&bt[(wn * 64 + nt * 16 + ln) * 40 + 8 * half], &bt[(wn * 64 + nt * 16 + ln) * 40 + 16 + 8 * half]);
      c[nt] = WMMA_BF16(a, b, c[nt]);
    }
  }
  // scatter D-frags into q/k/v [B][H][N][64] bf16
#pragma unroll
  for (int nt = 0; nt < 4; nt++) {
    int cg    = n0 + wn * 64 + nt * 16 + ln;          // 0..1535
    int which = cg >> 9;
    int col   = cg & 511;
    int h     = col >> 6, dd = col & 63;
    unsigned short* dst = (which == 0) ? qws : (which == 1) ? kws : vws;
#pragma unroll
    for (int r = 0; r < 8; r++) {
      int m = m0 + wm * 16 + 8 * half + r;
      int b = m >> 11, n = m & 2047;
      dst[((size_t)(b * HEADS + h) * SEQ + n) * DHEAD + dd] = f2bf(c[nt][r]);
    }
  }
}

// =====================================================================
// Kernel 2: flash attention, one (b,h) x 128 query rows per block.
// K tile staged by TDM (tensor_load_to_lds, padded to stride 72),
// V tile staged by global_load_async_to_lds_b128 then transposed in LDS,
// P round-trips through per-wave LDS (C-layout -> A-layout).
// =====================================================================
__global__ __launch_bounds__(256) void flash_attn(const unsigned short* __restrict__ qws,
                                                  const unsigned short* __restrict__ kws,
                                                  const unsigned short* __restrict__ vws,
                                                  unsigned short* __restrict__ attn) {
  __shared__ __align__(16) unsigned short kt[64 * 72];        // K tile [key][d] (padded)
  __shared__ __align__(16) unsigned short vraw[64 * 72];      // V tile [key][d] (padded)
  __shared__ __align__(16) unsigned short vt[64 * 72];        // V^T tile [d][key]
  __shared__ __align__(16) unsigned short pl[8 * 16 * 72];    // per-wave P [row][key]
  const int tid = threadIdx.x;
  const int w = tid >> 5, l = tid & 31, half = l >> 4, ln = l & 15;
  const int bh = blockIdx.x, b = bh >> 3, h = bh & 7;
  const int q0 = blockIdx.y * 128 + w * 16;

  // Q A-fragments held in registers for the whole kernel (K = d = 64 -> 2 frags)
  const unsigned short* qrow = qws + ((size_t)bh * SEQ + q0 + ln) * DHEAD;
  v16bf aq[2];
#pragma unroll
  for (int kk = 0; kk < 2; kk++)
    aq[kk] = load_frag2(qrow + kk * 32 +      8 * half,
                        qrow + kk * 32 + 16 + 8 * half);

  v8f o[4] = {};
  float m[8], lsum[8];
#pragma unroll
  for (int r = 0; r < 8; r++) { m[r] = -1e30f; lsum[r] = 0.f; }
  unsigned short* pw = pl + w * 16 * 72;

  for (int jt = 0; jt < SEQ / 64; ++jt) {
    const int j0 = jt * 64;
    __syncthreads();   // previous iteration done reading kt / vt

    // ---- TDM: K tile (contiguous 8KB) -> LDS with 16B pad per 128B row ----
    if (w == 0) {
      unsigned long long ga =
          (unsigned long long)(kws + ((size_t)bh * SEQ + j0) * DHEAD);
      u32x4 g0;
      g0[0] = 1u;                                        // count = 1 (valid D#)
      g0[1] = lds_off(&kt[0]);                           // lds_addr
      g0[2] = (unsigned)(ga & 0xFFFFFFFFu);              // global_addr[31:0]
      g0[3] = (unsigned)((ga >> 32) & 0x1FFFFFFu)        // global_addr[56:32]
              | 0x80000000u;                             // type = 2 (image)
      u32x8 g1;
      g1[0] = (1u << 16)        // data_size = 2 bytes
            | (1u << 20)        // pad_enable
            | (4u << 22)        // pad_interval: 32 DWORDs (=128B row)
            | (3u << 25);       // pad_amount: 4 DWORDs (=16B)
      g1[1] = 4096u << 16;      // tensor_dim0[15:0] = 4096 elements
      g1[2] = 1u << 16;         // tensor_dim0[31:16]=0 | tensor_dim1[15:0]=1
      g1[3] = 4096u << 16;      // tensor_dim1[31:16]=0 | tile_dim0 = 4096
      g1[4] = 0;                // tile_dim1 = 0 (unused), tile_dim2 = 0
      g1[5] = 4096u;            // tensor_dim0_stride[31:0]
      g1[6] = 0;
      g1[7] = 0;
      asm volatile("tensor_load_to_lds %0, %1" :: "s"(g0), "s"(g1) : "memory");
    }

    // ---- async copy: V tile -> vraw (each thread 2 x 16B chunks) ----
#pragma unroll
    for (int i = 0; i < 2; i++) {
      int ch = tid + i * 256;                           // 0..511
      int key = ch >> 3, part = ch & 7;
      unsigned ldsa = lds_off(&vraw[key * 72 + part * 8]);
      unsigned goff = (unsigned)((((bh * SEQ + j0 + key) * DHEAD) + part * 8) * 2);
      asm volatile("global_load_async_to_lds_b128 %0, %1, %2"
                   :: "v"(ldsa), "v"(goff), "s"(vws) : "memory");
    }
    asm volatile("s_wait_asynccnt 0" ::: "memory");
    if (w == 0) __builtin_amdgcn_s_wait_tensorcnt(0);
    __syncthreads();   // kt + vraw visible to all waves

    // ---- transpose vraw [key][d] -> vt [d][key] (LDS -> LDS) ----
#pragma unroll
    for (int i = 0; i < 16; i++) {
      int idx = tid + i * 256;
      int key = idx >> 6, dd = idx & 63;
      vt[dd * 72 + key] = vraw[key * 72 + dd];
    }
    __syncthreads();   // vt visible

    // ---- S = Q . K^T (16 rows x 64 keys per wave), K frags from LDS ----
    v8f s[4] = {};
#pragma unroll
    for (int kk = 0; kk < 2; kk++) {
#pragma unroll
      for (int nt = 0; nt < 4; nt++) {
        v16bf bk = load_frag2(&kt[(nt * 16 + ln) * 72 + kk * 32 + 8 * half], &kt[(nt * 16 + ln) * 72 + kk * 32 + 16 + 8 * half]);
        s[nt] = WMMA_BF16(aq[kk], bk, s[nt]);
      }
    }

    // ---- online softmax; row r lives in c[r] at row 8*half+r across 16 lanes ----
#pragma unroll
    for (int r = 0; r < 8; r++) {
      float t0 = s[0][r] * ATT_SCALE, t1 = s[1][r] * ATT_SCALE;
      float t2 = s[2][r] * ATT_SCALE, t3 = s[3][r] * ATT_SCALE;
      float rmax = fmaxf(fmaxf(t0, t1), fmaxf(t2, t3));
#pragma unroll
      for (int off = 1; off < 16; off <<= 1)
        rmax = fmaxf(rmax, __shfl_xor(rmax, off, 32));
      float mn = fmaxf(m[r], rmax);
      float al = __expf(m[r] - mn);
      m[r] = mn;
      float p0 = __expf(t0 - mn), p1 = __expf(t1 - mn);
      float p2 = __expf(t2 - mn), p3 = __expf(t3 - mn);
      int row = 8 * half + r;
      pw[row * 72 +  0 + ln] = f2bf(p0);
      pw[row * 72 + 16 + ln] = f2bf(p1);
      pw[row * 72 + 32 + ln] = f2bf(p2);
      pw[row * 72 + 48 + ln] = f2bf(p3);
      float ps = p0 + p1 + p2 + p3;
#pragma unroll
      for (int off = 1; off < 16; off <<= 1)
        ps += __shfl_xor(ps, off, 32);
      lsum[r] = lsum[r] * al + ps;
      o[0][r] *= al; o[1][r] *= al; o[2][r] *= al; o[3][r] *= al;
    }
    asm volatile("s_wait_dscnt 0" ::: "memory");      // P stores -> P loads (same wave)

    // ---- O += P . V  (K = 64 keys -> 2 frags) ----
#pragma unroll
    for (int kk = 0; kk < 2; kk++) {
      const unsigned short* pr = pw + ln * 72 + kk * 32;
      v16bf ap = load_frag2(pr + 8 * half, pr + 16 + 8 * half);
#pragma unroll
      for (int nt = 0; nt < 4; nt++) {
        v16bf bv = load_frag2(&vt[(nt * 16 + ln) * 72 + kk * 32 + 8 * half], &vt[(nt * 16 + ln) * 72 + kk * 32 + 16 + 8 * half]);
        o[nt] = WMMA_BF16(ap, bv, o[nt]);
      }
    }
  }

  // ---- epilogue: O / l, write bf16 attn_out [B][N][DMODEL] ----
#pragma unroll
  for (int r = 0; r < 8; r++) {
    float inv = 1.0f / lsum[r];
    int n = q0 + 8 * half + r;
    unsigned short* dst = attn + ((size_t)(b * SEQ + n)) * DMODEL + h * DHEAD;
#pragma unroll
    for (int nt = 0; nt < 4; nt++)
      dst[nt * 16 + ln] = f2bf(o[nt][r] * inv);
  }
}

// =====================================================================
// Kernel 3: out = attn_out @ w_out + b_out  (bf16 A, fp32 weights/out)
// =====================================================================
__global__ __launch_bounds__(256) void out_gemm(const unsigned short* __restrict__ attn,
                                                const float* __restrict__ wout,
                                                const float* __restrict__ bout,
                                                float* __restrict__ out) {
  __shared__ __align__(16) unsigned short bt[128 * 40];
  const int tid = threadIdx.x;
  const int w = tid >> 5, l = tid & 31, half = l >> 4, ln = l & 15;
  const int wm = w >> 1, wn = w & 1;
  const int m0 = blockIdx.x * 64;
  const int n0 = blockIdx.y * 128;
  const unsigned short* aptr = attn + (size_t)(m0 + wm * 16 + ln) * DMODEL;

  v8f c[4] = {};
  for (int kc = 0; kc < DMODEL; kc += 32) {
    __syncthreads();
#pragma unroll
    for (int i = 0; i < 8; i++) {
      int idx = tid + i * 256;
      int kp = idx >> 7, n = idx & 127;
      float a0 = wout[(size_t)(kc + 2 * kp)     * DMODEL + n0 + n];
      float a1 = wout[(size_t)(kc + 2 * kp + 1) * DMODEL + n0 + n];
      *(unsigned*)&bt[n * 40 + 2 * kp] = pack2bf(a0, a1);
    }
    if (kc + 32 < DMODEL)
      __builtin_prefetch(wout + (size_t)(kc + 32) * DMODEL + n0 + (tid & 127), 0, 0);
    __syncthreads();

    v16bf a = load_frag2(aptr + kc +      8 * half,
                         aptr + kc + 16 + 8 * half);
#pragma unroll
    for (int nt = 0; nt < 4; nt++) {
      v16bf b = load_frag2(&bt[(wn * 64 + nt * 16 + ln) * 40 + 8 * half], &bt[(wn * 64 + nt * 16 + ln) * 40 + 16 + 8 * half]);
      c[nt] = WMMA_BF16(a, b, c[nt]);
    }
  }
#pragma unroll
  for (int nt = 0; nt < 4; nt++) {
    int cg = n0 + wn * 64 + nt * 16 + ln;
    float bias = bout[cg];
#pragma unroll
    for (int r = 0; r < 8; r++) {
      int mrow = m0 + wm * 16 + 8 * half + r;
      out[(size_t)mrow * DMODEL + cg] = c[nt][r] + bias;
    }
  }
}

// =====================================================================
extern "C" void kernel_launch(void* const* d_in, const int* in_sizes, int n_in,
                              void* d_out, int out_size, void* d_ws, size_t ws_size,
                              hipStream_t stream) {
  const float* x    = (const float*)d_in[0];
  const float* wqkv = (const float*)d_in[1];
  const float* wout = (const float*)d_in[2];
  const float* bout = (const float*)d_in[3];
  float*       out  = (float*)d_out;

  // workspace: q, k, v, attn_out -- each B*H*N*64 bf16 = 8 MB (32 MB total)
  const size_t QN = (size_t)BATCH * HEADS * SEQ * DHEAD;
  unsigned short* ws   = (unsigned short*)d_ws;
  unsigned short* qws  = ws;
  unsigned short* kws  = ws + QN;
  unsigned short* vws  = ws + 2 * QN;
  unsigned short* attn = ws + 3 * QN;

  qkv_gemm <<<dim3(128, 12), 256, 0, stream>>>(x, wqkv, qws, kws, vws);
  flash_attn<<<dim3(BATCH * HEADS, SEQ / 128), 256, 0, stream>>>(qws, kws, vws, attn);
  out_gemm <<<dim3(128, 4), 256, 0, stream>>>(attn, wout, bout, out);
}
